// HFPrefixMLP_86371792322896
// MI455X (gfx1250) — matrix operand order repaired and checked by reading, not compiled, not run on hardware
//
#include <hip/hip_runtime.h>
#include <cstdint>
#include <cstddef>

typedef unsigned short u16;
typedef __attribute__((ext_vector_type(16))) __bf16        v16bf;
typedef __attribute__((ext_vector_type(8)))  float         v8f;
typedef __attribute__((ext_vector_type(4)))  unsigned int  v4u;

union ABFrag { v16bf v; v4u q[2]; };

static constexpr int kT = 16384;   // B*S tokens
static constexpr int kD = 1024;
static constexpr int kF = 2048;
static constexpr int kE = 8;
static constexpr int kHS = kF + 8;   // LDS h row stride (halves)
static constexpr int kOS = kD + 4;   // LDS out row stride (floats)
static constexpr int kSmemBytes = 32 * kHS * 2 + 32 * kOS * 4 + 32 * kE * 4; // 264192 B

__device__ __forceinline__ u16 f2bf(float f) {
  unsigned u = __float_as_uint(f);
  unsigned r = u + 0x7fffu + ((u >> 16) & 1u);   // round-to-nearest-even
  return (u16)(r >> 16);
}

__device__ __forceinline__ v8f vzero() {
  v8f z = {0.f, 0.f, 0.f, 0.f, 0.f, 0.f, 0.f, 0.f};
  return z;
}

#define WMMA_BF16(A, B, C) \
  __builtin_amdgcn_wmma_f32_16x16x32_bf16(false, (A), false, (B), (short)0, (C), false, false)

struct __attribute__((aligned(16))) US8 { u16 h[8]; };

// ------------------------------------------------------------------
// Weight swizzle: f32 [E][R][C] row-major  ->  bf16 WMMA B-fragment blocks.
// Fragment (e, nt, kk) = B tile rows nt*16..+15 (N), cols kk*32..+31 (K),
// laid out as 32 lanes x 16 halves contiguous (1 KB):
//   dst[((e*R/16 + nt)*C/32 + kk)*512 + l*16 + i]
//     = w[e][nt*16 + (l&15)][kk*32 + (i>>3)*16 + (l>>4)*8 + (i&7)]
// R/16 and C/32 are powers of two here.
// ------------------------------------------------------------------
__global__ __launch_bounds__(256) void cvt_w_tiled(const float* __restrict__ w,
                                                   u16* __restrict__ dst,
                                                   int R, int C, int nchunk) {
  const int KK  = C >> 5;            // power of two
  const int cpe = (R * C) >> 4;      // chunks per expert, power of two
  int c  = blockIdx.x * blockDim.x + threadIdx.x;
  int st = gridDim.x * blockDim.x;
  for (; c < nchunk; c += st) {
    const int e  = c / cpe;
    const int r  = c & (cpe - 1);
    const int l  = r & 31;
    const int q  = r >> 5;
    const int kk = q & (KK - 1);
    const int nt = q / KK;
    const int row = nt * 16 + (l & 15);
    const int col = kk * 32 + (l >> 4) * 8;
    const float* s = w + ((size_t)e * R + row) * C + col;
    u16 h[16];
#pragma unroll
    for (int i = 0; i < 16; ++i) h[i] = f2bf(s[(i >> 3) * 16 + (i & 7)]);
    US8* d8 = (US8*)(dst + (size_t)c * 16);
    US8 o0 = {{h[0], h[1], h[2], h[3], h[4], h[5], h[6], h[7]}};
    US8 o1 = {{h[8], h[9], h[10], h[11], h[12], h[13], h[14], h[15]}};
    d8[0] = o0; d8[1] = o1;
  }
}

// ------------------------------------------------------------------
// x swizzle: f32 [T][D] -> bf16 WMMA A-fragment blocks.
// Fragment (tt, mt, kk) = A tile tokens tt*32+mt*16..+15 (M), cols kk*32..+31 (K):
//   halves 0..7  <- K = kk*32 +      (l>>4)*8 + j   (ISA A layout, group 0)
//   halves 8..15 <- K = kk*32 + 16 + (l>>4)*8 + j   (group 1)
//   dst[(((tt*2+mt)*32 + kk)*32 + l)*16 + i]
// ------------------------------------------------------------------
__global__ __launch_bounds__(256) void cvt_x_tiled(const float* __restrict__ x,
                                                   u16* __restrict__ dst, int nchunk) {
  int c  = blockIdx.x * blockDim.x + threadIdx.x;
  int st = gridDim.x * blockDim.x;
  for (; c < nchunk; c += st) {
    const int l  = c & 31;
    const int kk = (c >> 5) & 31;
    const int mt = (c >> 10) & 1;
    const int tt = c >> 11;
    const int tok = tt * 32 + mt * 16 + (l & 15);
    const int d0  = kk * 32 + (l >> 4) * 8;
    const float* s = x + (size_t)tok * kD;
    u16 h[16];
#pragma unroll
    for (int i = 0; i < 8; ++i) h[i] = f2bf(s[d0 + i]);
#pragma unroll
    for (int i = 0; i < 8; ++i) h[8 + i] = f2bf(s[d0 + 16 + i]);
    US8* d8 = (US8*)(dst + (size_t)c * 16);
    US8 o0 = {{h[0], h[1], h[2], h[3], h[4], h[5], h[6], h[7]}};
    US8 o1 = {{h[8], h[9], h[10], h[11], h[12], h[13], h[14], h[15]}};
    d8[0] = o0; d8[1] = o1;
  }
}

// ---------------- router: logits -> top-2 renormalized combine weights ----------------
__global__ __launch_bounds__(256) void router_topk(const float* __restrict__ x,
                                                   const float* __restrict__ rw,
                                                   float* __restrict__ cw) {
  __shared__ float s_rw[kE * kD];   // 32 KB
  const int tid = threadIdx.x;
  for (int i = tid; i < kE * kD; i += 256) s_rw[i] = rw[i];
  __syncthreads();

  const int tok = blockIdx.x * 16 + (tid >> 4);
  const int c   = tid & 15;
  float acc[kE];
#pragma unroll
  for (int e = 0; e < kE; ++e) acc[e] = 0.f;
  const float* xrow = x + (size_t)tok * kD;
  for (int d = c; d < kD; d += 16) {
    float xv = xrow[d];
#pragma unroll
    for (int e = 0; e < kE; ++e) acc[e] += xv * s_rw[e * kD + d];
  }
#pragma unroll
  for (int e = 0; e < kE; ++e) {
#pragma unroll
    for (int m = 8; m >= 1; m >>= 1) acc[e] += __shfl_xor(acc[e], m, 16);
  }
  if (c == 0) {
    int i1 = 0; float v1 = acc[0];
#pragma unroll
    for (int e = 1; e < kE; ++e) if (acc[e] > v1) { v1 = acc[e]; i1 = e; }
    int i2 = -1; float v2 = -3.4e38f;
#pragma unroll
    for (int e = 0; e < kE; ++e) if (e != i1 && acc[e] > v2) { v2 = acc[e]; i2 = e; }
    float p1 = 1.f / (1.f + __expf(v2 - v1));
    float p2 = 1.f - p1;
    float* o = cw + (size_t)tok * kE;
#pragma unroll
    for (int e = 0; e < kE; ++e) o[e] = (e == i1) ? p1 : ((e == i2) ? p2 : 0.f);
  }
}

// ---------------- fused MoE over pre-swizzled fragments ----------------
__global__ __launch_bounds__(256) void moe_fused(const u16* __restrict__ xt,
                                                 const u16* __restrict__ wgt,
                                                 const u16* __restrict__ wut,
                                                 const u16* __restrict__ wdt,
                                                 const float* __restrict__ cw,
                                                 float* __restrict__ out) {
  extern __shared__ char smem[];
  u16*   sh_h   = (u16*)smem;
  float* sh_out = (float*)(smem + 32 * kHS * 2);
  float* sh_cw  = (float*)(smem + 32 * kHS * 2 + 32 * kOS * 4);

  const int tid  = threadIdx.x;
  const int wv   = tid >> 5;
  const int lane = tid & 31;
  const int nrow = lane & 15;
  const int sel  = lane >> 4;
  const int tok0 = blockIdx.x * 32;

  for (int i = tid; i < 32 * kOS; i += 256) sh_out[i] = 0.f;
  sh_cw[tid] = cw[(size_t)tok0 * kE + tid];
  __syncthreads();

  // A fragment streams for this token tile (mt = 0,1); advance 512 halves per kk.
  const u16* xa0 = xt + (size_t)(blockIdx.x * 2 + 0) * 16384 + lane * 16;
  const u16* xa1 = xt + (size_t)(blockIdx.x * 2 + 1) * 16384 + lane * 16;

  for (int e = 0; e < kE; ++e) {
    bool any = false;
    for (int r = 0; r < 32; ++r) any = any || (sh_cw[r * kE + e] != 0.f);
    if (!any) continue;

    // ---- phase 1: gate/up GEMMs + SwiGLU -> h in LDS ----
    const u16* wgt_e = wgt + (size_t)e * kF * kD;
    const u16* wut_e = wut + (size_t)e * kF * kD;
    for (int p = 0; p < 8; ++p) {                    // 2 n-tiles per iteration
      const int nt0 = wv * 16 + 2 * p;
      const u16* bg0 = wgt_e + (size_t)(nt0 + 0) * 16384 + lane * 16;
      const u16* bg1 = wgt_e + (size_t)(nt0 + 1) * 16384 + lane * 16;
      const u16* bu0 = wut_e + (size_t)(nt0 + 0) * 16384 + lane * 16;
      const u16* bu1 = wut_e + (size_t)(nt0 + 1) * 16384 + lane * 16;
      v8f ag[2][2], au[2][2];                        // [mt][nti]
#pragma unroll
      for (int m = 0; m < 2; ++m)
#pragma unroll
        for (int n = 0; n < 2; ++n) { ag[m][n] = vzero(); au[m][n] = vzero(); }
#pragma unroll 4
      for (int kk = 0; kk < 32; ++kk) {
        const int off = kk * 512;
        __builtin_prefetch(bg0 + off + 4096, 0, 1);
        __builtin_prefetch(bu0 + off + 4096, 0, 1);
        ABFrag A0, A1, Bg0, Bg1, Bu0, Bu1;
        A0.q[0]  = *(const v4u*)(xa0 + off); A0.q[1]  = *(const v4u*)(xa0 + off + 8);
        A1.q[0]  = *(const v4u*)(xa1 + off); A1.q[1]  = *(const v4u*)(xa1 + off + 8);
        Bg0.q[0] = *(const v4u*)(bg0 + off); Bg0.q[1] = *(const v4u*)(bg0 + off + 8);
        Bg1.q[0] = *(const v4u*)(bg1 + off); Bg1.q[1] = *(const v4u*)(bg1 + off + 8);
        Bu0.q[0] = *(const v4u*)(bu0 + off); Bu0.q[1] = *(const v4u*)(bu0 + off + 8);
        Bu1.q[0] = *(const v4u*)(bu1 + off); Bu1.q[1] = *(const v4u*)(bu1 + off + 8);
        ag[0][0] = WMMA_BF16(A0.v, Bg0.v, ag[0][0]);
        ag[1][0] = WMMA_BF16(A1.v, Bg0.v, ag[1][0]);
        ag[0][1] = WMMA_BF16(A0.v, Bg1.v, ag[0][1]);
        ag[1][1] = WMMA_BF16(A1.v, Bg1.v, ag[1][1]);
        au[0][0] = WMMA_BF16(A0.v, Bu0.v, au[0][0]);
        au[1][0] = WMMA_BF16(A1.v, Bu0.v, au[1][0]);
        au[0][1] = WMMA_BF16(A0.v, Bu1.v, au[0][1]);
        au[1][1] = WMMA_BF16(A1.v, Bu1.v, au[1][1]);
      }
#pragma unroll
      for (int n = 0; n < 2; ++n) {
        const int fb = (nt0 + n) * 16;
#pragma unroll
        for (int j = 0; j < 8; ++j) {
          const int r = sel * 8 + j;
          float g = ag[0][n][j], u = au[0][n][j];
          sh_h[r * kHS + fb + nrow] = f2bf((g / (1.f + __expf(-g))) * u);
          g = ag[1][n][j]; u = au[1][n][j];
          sh_h[(16 + r) * kHS + fb + nrow] = f2bf((g / (1.f + __expf(-g))) * u);
        }
      }
    }
    __syncthreads();

    float cw0[8], cw1[8];
#pragma unroll
    for (int j = 0; j < 8; ++j) {
      cw0[j] = sh_cw[(sel * 8 + j) * kE + e];
      cw1[j] = sh_cw[(16 + sel * 8 + j) * kE + e];
    }

    // ---- phase 2: down GEMM, weighted accumulate into LDS out tile ----
    const u16* wdt_e = wdt + (size_t)e * kD * kF;
    const u16* h0row = sh_h + (size_t)nrow * kHS;
    const u16* h1row = sh_h + (size_t)(16 + nrow) * kHS;
    for (int p = 0; p < 4; ++p) {                    // 2 d-tiles per iteration
      const int dt0 = wv * 8 + 2 * p;
      const u16* bd0 = wdt_e + (size_t)(dt0 + 0) * 32768 + lane * 16;
      const u16* bd1 = wdt_e + (size_t)(dt0 + 1) * 32768 + lane * 16;
      v8f ac[2][2];                                  // [mt][dti]
#pragma unroll
      for (int m = 0; m < 2; ++m)
#pragma unroll
        for (int n = 0; n < 2; ++n) ac[m][n] = vzero();
#pragma unroll 4
      for (int kk = 0; kk < 64; ++kk) {
        const int f0 = kk * 32, off = kk * 512;
        __builtin_prefetch(bd0 + off + 4096, 0, 1);
        ABFrag A0, A1, B0, B1;
        A0.q[0] = *(const v4u*)(h0row + f0 + sel * 8);
        A0.q[1] = *(const v4u*)(h0row + f0 + 16 + sel * 8);
        A1.q[0] = *(const v4u*)(h1row + f0 + sel * 8);
        A1.q[1] = *(const v4u*)(h1row + f0 + 16 + sel * 8);
        B0.q[0] = *(const v4u*)(bd0 + off); B0.q[1] = *(const v4u*)(bd0 + off + 8);
        B1.q[0] = *(const v4u*)(bd1 + off); B1.q[1] = *(const v4u*)(bd1 + off + 8);
        ac[0][0] = WMMA_BF16(A0.v, B0.v, ac[0][0]);
        ac[1][0] = WMMA_BF16(A1.v, B0.v, ac[1][0]);
        ac[0][1] = WMMA_BF16(A0.v, B1.v, ac[0][1]);
        ac[1][1] = WMMA_BF16(A1.v, B1.v, ac[1][1]);
      }
#pragma unroll
      for (int n = 0; n < 2; ++n) {
        const int db = (dt0 + n) * 16;
#pragma unroll
        for (int j = 0; j < 8; ++j) {
          const int r = sel * 8 + j;
          sh_out[r * kOS + db + nrow]        += cw0[j] * ac[0][n][j];
          sh_out[(16 + r) * kOS + db + nrow] += cw1[j] * ac[1][n][j];
        }
      }
    }
    __syncthreads();
  }

  for (int i = tid; i < 32 * kD; i += 256) {
    const int r = i >> 10, d = i & (kD - 1);
    out[(size_t)(tok0 + r) * kD + d] = sh_out[r * kOS + d];
  }
}

// ---------------- host launch ----------------
extern "C" void kernel_launch(void* const* d_in, const int* in_sizes, int n_in,
                              void* d_out, int out_size, void* d_ws, size_t ws_size,
                              hipStream_t stream) {
  (void)in_sizes; (void)n_in; (void)out_size; (void)ws_size;
  const float* x  = (const float*)d_in[0];   // [B,S,D]
  const float* rw = (const float*)d_in[1];   // [E,D]
  const float* wg = (const float*)d_in[2];   // [E,F,D]
  const float* wu = (const float*)d_in[3];   // [E,F,D]
  const float* wd = (const float*)d_in[4];   // [E,D,F]
  float* out = (float*)d_out;

  char* ws = (char*)d_ws;
  size_t off = 0;
  float* cw  = (float*)(ws + off); off += (size_t)kT * kE * sizeof(float);
  u16*   xt  = (u16*)(ws + off);   off += (size_t)kT * kD * sizeof(u16);
  u16*   wgt = (u16*)(ws + off);   off += (size_t)kE * kF * kD * sizeof(u16);
  u16*   wut = (u16*)(ws + off);   off += (size_t)kE * kF * kD * sizeof(u16);
  u16*   wdt = (u16*)(ws + off);   off += (size_t)kE * kD * kF * sizeof(u16);

  cvt_x_tiled<<<2048, 256, 0, stream>>>(x, xt, kT * kD / 16);
  cvt_w_tiled<<<2048, 256, 0, stream>>>(wg, wgt, kF, kD, kE * kF * kD / 16);
  cvt_w_tiled<<<2048, 256, 0, stream>>>(wu, wut, kF, kD, kE * kF * kD / 16);
  cvt_w_tiled<<<2048, 256, 0, stream>>>(wd, wdt, kD, kF, kE * kD * kF / 16);

  router_topk<<<kT / 16, 256, 0, stream>>>(x, rw, cw);

  (void)hipFuncSetAttribute(reinterpret_cast<const void*>(moe_fused),
                            hipFuncAttributeMaxDynamicSharedMemorySize, kSmemBytes);
  moe_fused<<<kT / 32, 256, kSmemBytes, stream>>>(xt, wgt, wut, wdt, cw, out);
}
